// GaborTransform_64055142252567
// MI455X (gfx1250) — matrix operand order repaired and checked by reading, not compiled, not run on hardware
//
#include <hip/hip_runtime.h>
#include <hip/hip_bf16.h>
#include <math.h>

// Gabor transform on MI455X (gfx1250):
//   result[b,f,t] = sum_l signal[b,l] * exp(-0.5*((l-tc[t])/819.2)^2) * cos(2*pi*fv[f]*l/8192)
// Stage 1: precompute fp16 tables in d_ws: window[128,8192] (2 MB), cos[128,8192] (2 MB),
//          and fp16 copy of signal [32,8192] (512 KB).  Workspace requirement: 4.5 MB.
// Stage 2: per-batch GEMM [T,L]x[L,F] via v_wmma_f32_16x16x32_f16, f32 accumulation.
//          Each wave owns one 16t x 32f strip (2 accumulators) so every A fragment
//          feeds two WMMAs: per-WMMA cost ~4 v_pk_mul_f16 + 3 b128 loads.

typedef __attribute__((ext_vector_type(16))) _Float16 v16h;
typedef __attribute__((ext_vector_type(8)))  _Float16 v8h;
typedef __attribute__((ext_vector_type(8)))  float    v8f;

#define GAB_L   8192
#define GAB_NT  128
#define GAB_NF  128
#define GAB_NB  32
#define GAB_SIGW 819.2f   // 0.1 * 8192

// ---------------------------------------------------------------------------
// Stage 1: rows 0..127 -> window[t][l]; rows 128..255 -> cos[f][l];
//          rows 256..287 -> fp16 signal copy.
// cos phase reduced exactly in integers: (fv*l) mod 8192 -> angle in [0,2*pi).
// ---------------------------------------------------------------------------
__global__ void __launch_bounds__(256)
gabor_tables_kernel(const float* __restrict__ sig,
                    _Float16* __restrict__ win,
                    _Float16* __restrict__ cosm,
                    _Float16* __restrict__ sigh) {
    const int idx = blockIdx.x * 256 + threadIdx.x;   // 0 .. 288*8192-1
    const int r = idx >> 13;                          // row 0..287
    const int l = idx & (GAB_L - 1);
    if (r < GAB_NT) {
        // time_centers = trunc(linspace(0, L-1, NT)) -> floor since nonnegative
        const float tc = (float)(int)((float)r * (8191.0f / 127.0f));
        const float z  = ((float)l - tc) * (1.0f / GAB_SIGW);
        win[r * GAB_L + l] = (_Float16)__expf(-0.5f * z * z);
    } else if (r < 2 * GAB_NT) {
        const int f = r - GAB_NT;
        // freq_vals = trunc(linspace(0, L/2, NF))
        const int fv = (int)((float)f * (4096.0f / 127.0f));
        const int m  = (fv * l) & (GAB_L - 1);        // exact phase mod L (fits in i32)
        const float ang = 6.28318530717958647692f * (float)m * (1.0f / (float)GAB_L);
        cosm[f * GAB_L + l] = (_Float16)cosf(ang);
    } else {
        const int bi = r - 2 * GAB_NT;                // batch 0..31
        sigh[bi * GAB_L + l] = (_Float16)sig[bi * GAB_L + l];
    }
}

// ---------------------------------------------------------------------------
// Stage 2: WMMA GEMM. Block (256 thr = 8 waves) covers (b, 2 t-tiles):
//   wave w: t_tile = 2*(blockIdx&3) + (w>>2), f-group = (w&3) -> f in [32*fg, 32*fg+32)
// Fragment layouts (CDNA5 ISA, wave32):
//  A 16x32 f16 : lane lo = row M; half h -> K = hi*16+h (order chosen here, same as B)
//  B 32x16 f16 : lane lo = col N; half h -> K = hi*16+h (order chosen here, same as A)
//  D 16x16 f32 : VGPR i -> M = i + 8*hi ; lane lo -> N
// ---------------------------------------------------------------------------
__global__ void __launch_bounds__(256)
gabor_wmma_kernel(const _Float16* __restrict__ sigh,
                  const _Float16* __restrict__ win,
                  const _Float16* __restrict__ cosm,
                  float* __restrict__ out) {
    const int wave = threadIdx.x >> 5;                 // 0..7
    const int b    = blockIdx.x >> 2;                  // 0..31
    const int t0   = (2 * (blockIdx.x & 3) + (wave >> 2)) * 16;  // t-tile origin
    const int f0   = (wave & 3) * 32;                  // f-strip origin (2 tiles)
    const int lane = threadIdx.x & 31;
    const int lo   = lane & 15;
    const int hi   = lane >> 4;

    const _Float16* sigb = sigh + b * GAB_L;
    const _Float16* winr = win  + (t0 + lo) * GAB_L;          // A row (M = lo)
    const _Float16* cr0  = cosm + (f0 + lo) * GAB_L;          // B col, tile 0
    const _Float16* cr1  = cosm + (f0 + 16 + lo) * GAB_L;     // B col, tile 1

    v8f acc0 = {};
    v8f acc1 = {};

#pragma unroll 2
    for (int k0 = 0; k0 < GAB_L; k0 += 32) {
        const int ab = k0 + hi * 16;    // base of this lane's two 8-half A runs
        // fp16 signal + window: 16B vector loads each
        v8h s0 = *(const v8h*)(sigb + ab);
        v8h s1 = *(const v8h*)(sigb + ab + 8);
        v8h w0 = *(const v8h*)(winr + ab);
        v8h w1 = *(const v8h*)(winr + ab + 8);
        // B fragments: one contiguous 32B fp16 run per lane per tile
        v16h b0 = *(const v16h*)(cr0 + k0 + hi * 16);
        v16h b1 = *(const v16h*)(cr1 + k0 + hi * 16);

        // A fragment: signal * window, pure packed f16 muls (no cvt in loop)
        v16h a;
#pragma unroll
        for (int i = 0; i < 8; ++i) {
            a[i]     = s0[i] * w0[i];
            a[i + 8] = s1[i] * w1[i];
        }

        // One A feeds two WMMAs (f-tile 0 and 1), f32 accumulation
        acc0 = __builtin_amdgcn_wmma_f32_16x16x32_f16(
                   false, a, false, b0, (short)0, acc0, false, false);
        acc1 = __builtin_amdgcn_wmma_f32_16x16x32_f16(
                   false, a, false, b1, (short)0, acc1, false, false);
    }

    // Store: lane holds D[M = i+8*hi, N = lo] -> out[b][f][t0+M]; 8 floats per lane
    // are contiguous in t -> single 32B aligned vector store per tile.
    float* o0 = out + ((size_t)(b * GAB_NF + f0 + lo) * GAB_NT) + t0 + hi * 8;
    *(v8f*)o0 = acc0;
    *(v8f*)(o0 + 16 * GAB_NT) = acc1;
}

extern "C" void kernel_launch(void* const* d_in, const int* in_sizes, int n_in,
                              void* d_out, int out_size, void* d_ws, size_t ws_size,
                              hipStream_t stream) {
    const float* sig = (const float*)d_in[0];
    float* out = (float*)d_out;

    // Workspace: window (2 MB) | cos (2 MB) | fp16 signal (512 KB)
    _Float16* win  = (_Float16*)d_ws;
    _Float16* cosm = win  + (size_t)GAB_NT * GAB_L;
    _Float16* sigh = cosm + (size_t)GAB_NF * GAB_L;

    // Stage 1: 288 rows x 8192 cols, one element per thread.
    gabor_tables_kernel<<<(288 * GAB_L) / 256, 256, 0, stream>>>(sig, win, cosm, sigh);

    // Stage 2: 32 b x 4 t-pairs blocks; 8 waves/block = 2 t-tiles x 4 f-strips.
    gabor_wmma_kernel<<<GAB_NB * 4, 256, 0, stream>>>(sigh, win, cosm, out);
}
